// DynamicJKGCN_21689584844949
// MI455X (gfx1250) — matrix operand order repaired and checked by reading, not compiled, not run on hardware
//
#include <hip/hip_runtime.h>
#include <hip/hip_bf16.h>
#include <math.h>

#define N_NODES 50000
#define N_EDGES 800000
#define F_IN    256
#define F_H     128
#define F_OUT   64

typedef __attribute__((ext_vector_type(16))) _Float16 v16h;
typedef __attribute__((ext_vector_type(8)))  float    v8f;

// ---------------- utility kernels ----------------

__global__ void k_fill(float* __restrict__ p, float v, int n) {
  int i = blockIdx.x * blockDim.x + threadIdx.x;
  if (i < n) p[i] = v;
}

__global__ void k_cvt_f16(const float* __restrict__ s, _Float16* __restrict__ d, int n) {
  int i = blockIdx.x * blockDim.x + threadIdx.x;
  if (i < n) d[i] = (_Float16)s[i];
}

__global__ void k_edge_deg(const int* __restrict__ row, const int* __restrict__ col,
                           float* __restrict__ deg, float* __restrict__ odeg, int E) {
  int e = blockIdx.x * blockDim.x + threadIdx.x;
  if (e < E) {
    atomicAdd(&deg[col[e]], 1.0f);   // in-degree over dst (self-loop folded into init=1)
    atomicAdd(&odeg[row[e]], 1.0f);  // out-degree over src
  }
}

__global__ void k_dinv(const float* __restrict__ deg, float* __restrict__ dinv, int n) {
  int i = blockIdx.x * blockDim.x + threadIdx.x;
  if (i < n) dinv[i] = rsqrtf(fmaxf(deg[i], 1.0f));
}

__global__ void k_softmax3(const float* __restrict__ jw, float* __restrict__ w) {
  if (blockIdx.x == 0 && threadIdx.x == 0) {
    float a = jw[0], b = jw[1], c = jw[2];
    float m = fmaxf(a, fmaxf(b, c));
    float ea = expf(a - m), eb = expf(b - m), ec = expf(c - m);
    float s = ea + eb + ec;
    w[0] = ea / s; w[1] = eb / s; w[2] = ec / s;
  }
}

// Pack W (K x Nc, row-major f32) into the CDNA5 WMMA f16 B-operand layout:
// tile t = tn*KT + kt holds the 32x16 slice; lane l -> column n = tn*16 + (l&15),
// element e (0..15) -> K row = 32*kt + (e&7) + 8*(l>>4) + 16*(e>>3)   (ISA 7.12.2, B rows across lanes)
__global__ void k_pack_w(const float* __restrict__ W, _Float16* __restrict__ out,
                         int K, int Nc) {
  int tid = blockIdx.x * blockDim.x + threadIdx.x;
  int KT = K >> 5, NT = Nc >> 4;
  if (tid >= KT * NT * 32) return;
  int l = tid & 31;
  int t = tid >> 5;               // t = tn*KT + kt
  int kt = t % KT;
  int half = l >> 4;
  int n = ((t / KT) << 4) + (l & 15);
  _Float16* o = out + ((size_t)t * 32 + l) * 16;
#pragma unroll
  for (int e = 0; e < 16; ++e) {
    int k = 32 * kt + (e & 7) + 8 * half + 16 * (e >> 3);
    o[e] = (_Float16)W[(size_t)k * Nc + n];
  }
}

// ---------------- WMMA GEMM: out[M x Nc] = A[M x K] (f16) @ Bpack (+bias) ----------------
// One wave computes one 16x16 tile of D using v_wmma_f32_16x16x32_f16 along K.
// A-operand layout (16-bit A 16x32, ISA 7.12.2): lane l -> row m = l&15, half = l>>4,
// element e -> K = (e&7) + 8*half + 16*(e>>3)  => two contiguous 16B runs at +0 and +16.
__global__ __launch_bounds__(128) void k_gemm_wmma(
    const _Float16* __restrict__ A, const _Float16* __restrict__ Bp,
    float* __restrict__ out, const float* __restrict__ bias,
    int M, int K, int Nc) {
  const int lane = threadIdx.x & 31;
  const int wave = blockIdx.x * (blockDim.x >> 5) + (threadIdx.x >> 5);
  const int NT = Nc >> 4;
  const int KT = K >> 5;
  const int Mtiles = (M + 15) >> 4;
  if (wave >= Mtiles * NT) return;            // wave-uniform exit: EXEC stays all-ones
  const int tn = wave % NT;
  const int tm = wave / NT;
  const int half = lane >> 4;
  const int lmod = lane & 15;
  int row = tm * 16 + lmod;
  if (row >= M) row = M - 1;                  // clamp tail rows (stores are guarded)
  const _Float16* arow  = A + (size_t)row * K + 8 * half;
  const _Float16* btile = Bp + (size_t)(tn * KT) * 512 + lane * 16;
  v8f c = {};
#pragma unroll 4
  for (int kt = 0; kt < KT; ++kt) {
    v16h a, b;
    ((float4*)&a)[0] = *(const float4*)(arow + kt * 32);        // K = 8h + 0..7
    ((float4*)&a)[1] = *(const float4*)(arow + kt * 32 + 16);   // K = 16 + 8h + 0..7
    ((float4*)&b)[0] = ((const float4*)(btile + (size_t)kt * 512))[0];
    ((float4*)&b)[1] = ((const float4*)(btile + (size_t)kt * 512))[1];
    c = __builtin_amdgcn_wmma_f32_16x16x32_f16(false, a, false, b, (short)0, c,
                                               false, false);
  }
  const int colg = tn * 16 + lmod;
  const float bv = bias ? bias[colg] : 0.0f;
#pragma unroll
  for (int r = 0; r < 8; ++r) {               // D layout: VGPR r -> row r + 8*half
    int rg = tm * 16 + half * 8 + r;
    if (rg < M) out[(size_t)rg * Nc + colg] = c[r] + bv;
  }
}

// ---------------- graph aggregation kernels (H = 128 hard-coded) ----------------

// self-loop term: agg = dinv[v]^2 * t
__global__ void k_self_scale(const float* __restrict__ t, const float* __restrict__ dinv,
                             float* __restrict__ agg, int total) {
  int i = blockIdx.x * blockDim.x + threadIdx.x;
  if (i < total) {
    float dv = dinv[i >> 7];
    agg[i] = dv * dv * t[i];
  }
}

// agg[col[e]] += dinv[row]*dinv[col] * t[row[e]]   (4 feats / thread)
__global__ void k_scatter_gcn(const int* __restrict__ row, const int* __restrict__ col,
                              const float* __restrict__ dinv, const float* __restrict__ t,
                              float* __restrict__ agg, int total) {
  int idx = blockIdx.x * blockDim.x + threadIdx.x;
  if (idx >= total) return;
  int e = idx >> 5;
  int c = (idx & 31) << 2;
  int s = row[e], d = col[e];
  float sc = dinv[s] * dinv[d];
  float4 hv = *(const float4*)(t + (size_t)s * 128 + c);
  float* o = agg + (size_t)d * 128 + c;
  atomicAdd(o + 0, sc * hv.x);
  atomicAdd(o + 1, sc * hv.y);
  atomicAdd(o + 2, sc * hv.z);
  atomicAdd(o + 3, sc * hv.w);
}

// nsum[row[e]] += h[col[e]]
__global__ void k_scatter_mean(const int* __restrict__ row, const int* __restrict__ col,
                               const float* __restrict__ h, float* __restrict__ agg,
                               int total) {
  int idx = blockIdx.x * blockDim.x + threadIdx.x;
  if (idx >= total) return;
  int e = idx >> 5;
  int c = (idx & 31) << 2;
  float4 hv = *(const float4*)(h + (size_t)col[e] * 128 + c);
  float* o = agg + (size_t)row[e] * 128 + c;
  atomicAdd(o + 0, hv.x);
  atomicAdd(o + 1, hv.y);
  atomicAdd(o + 2, hv.z);
  atomicAdd(o + 3, hv.w);
}

// out = relu(agg + b)
__global__ void k_bias_relu(const float* __restrict__ agg, const float* __restrict__ b,
                            float* __restrict__ out, int total) {
  int i = blockIdx.x * blockDim.x + threadIdx.x;
  if (i < total) {
    float v = agg[i] + b[i & 127];
    out[i] = v > 0.0f ? v : 0.0f;
  }
}

// layer-0 tail: neighbor mean + start JK accumulation
__global__ void k_combine0(const float* __restrict__ h0pre, const float* __restrict__ nsum,
                           const float* __restrict__ odeg, _Float16* __restrict__ hh,
                           float* __restrict__ jk, const float* __restrict__ w, int total) {
  int i = blockIdx.x * blockDim.x + threadIdx.x;
  if (i < total) {
    float od = odeg[i >> 7];
    float h = h0pre[i];
    if (od > 0.0f) h = nsum[i] / od;
    hh[i] = (_Float16)h;
    jk[i] = w[0] * h;
  }
}

// layers 1/2 tail: h = relu(agg+b); hh = f16(h); jk += w[widx]*h
__global__ void k_combine(const float* __restrict__ agg, const float* __restrict__ b,
                          _Float16* __restrict__ hh, float* __restrict__ jk,
                          const float* __restrict__ w, int widx, int total) {
  int i = blockIdx.x * blockDim.x + threadIdx.x;
  if (i < total) {
    float h = agg[i] + b[i & 127];
    h = h > 0.0f ? h : 0.0f;
    hh[i] = (_Float16)h;
    jk[i] += w[widx] * h;
  }
}

// ---------------- host orchestration ----------------

extern "C" void kernel_launch(void* const* d_in, const int* in_sizes, int n_in,
                              void* d_out, int out_size, void* d_ws, size_t ws_size,
                              hipStream_t stream) {
  const float* x   = (const float*)d_in[0];
  const int*   ei  = (const int*)d_in[1];
  const int*   row = ei;             // src
  const int*   col = ei + N_EDGES;   // dst
  const float* W1  = (const float*)d_in[2];
  const float* b1  = (const float*)d_in[3];
  const float* W2  = (const float*)d_in[4];
  const float* b2  = (const float*)d_in[5];
  const float* W3  = (const float*)d_in[6];
  const float* b3  = (const float*)d_in[7];
  const float* jw  = (const float*)d_in[8];
  const float* Wfc = (const float*)d_in[9];
  const float* bfc = (const float*)d_in[10];
  float* out = (float*)d_out;

  char* base = (char*)d_ws;
  size_t off = 0;
  auto alloc = [&](size_t bytes) {
    char* p = base + off;
    off += (bytes + 255) & ~(size_t)255;
    return p;
  };
  float*    deg  = (float*)alloc((size_t)N_NODES * 4);
  float*    odeg = (float*)alloc((size_t)N_NODES * 4);
  float*    dinv = (float*)alloc((size_t)N_NODES * 4);
  float*    wsm  = (float*)alloc(256);                        // softmax(jk_weights)
  _Float16* w1p  = (_Float16*)alloc((size_t)F_IN * F_H * 2);
  _Float16* w2p  = (_Float16*)alloc((size_t)F_H * F_H * 2);
  _Float16* w3p  = (_Float16*)alloc((size_t)F_H * F_H * 2);
  _Float16* wfp  = (_Float16*)alloc((size_t)F_H * F_OUT * 2);
  _Float16* xh   = (_Float16*)alloc((size_t)N_NODES * F_IN * 2);
  float*    t    = (float*)alloc((size_t)N_NODES * F_H * 4);  // GEMM out / h0pre / jk_f16
  float*    agg  = (float*)alloc((size_t)N_NODES * F_H * 4);  // aggregation accumulator
  _Float16* hh   = (_Float16*)alloc((size_t)N_NODES * F_H * 2); // current layer in f16
  float*    jk   = (float*)alloc((size_t)N_NODES * F_H * 4);  // JK accumulator

  const int NH = N_NODES * F_H;          // 6.4M
  const int E32 = N_EDGES * 32;          // 25.6M edge-feature chunks
  dim3 blk(256);
  auto grid1 = [](int n) { return dim3((unsigned)((n + 255) / 256)); };
  auto gemm = [&](const _Float16* A, const _Float16* Bp, float* O, const float* bias,
                  int M, int K, int Nc) {
    int waves = ((M + 15) / 16) * (Nc / 16);
    k_gemm_wmma<<<dim3((unsigned)((waves + 3) / 4)), dim3(128), 0, stream>>>(A, Bp, O,
                                                                             bias, M, K, Nc);
  };
  auto pack = [&](const float* W, _Float16* P, int K, int Nc) {
    int n = (K / 32) * (Nc / 16) * 32;
    k_pack_w<<<grid1(n), blk, 0, stream>>>(W, P, K, Nc);
  };

  // degrees / norms / softmax weights
  k_fill<<<grid1(N_NODES), blk, 0, stream>>>(deg, 1.0f, N_NODES);   // self-loop
  k_fill<<<grid1(N_NODES), blk, 0, stream>>>(odeg, 0.0f, N_NODES);
  k_edge_deg<<<grid1(N_EDGES), blk, 0, stream>>>(row, col, deg, odeg, N_EDGES);
  k_dinv<<<grid1(N_NODES), blk, 0, stream>>>(deg, dinv, N_NODES);
  k_softmax3<<<dim3(1), dim3(32), 0, stream>>>(jw, wsm);

  // f16 conversions / weight packing
  k_cvt_f16<<<grid1(N_NODES * F_IN), blk, 0, stream>>>(x, xh, N_NODES * F_IN);
  pack(W1, w1p, F_IN, F_H);
  pack(W2, w2p, F_H, F_H);
  pack(W3, w3p, F_H, F_H);
  pack(Wfc, wfp, F_H, F_OUT);

  // ---- layer 0: h0pre = relu(Agg(x@W1) + b1); h0 = out-neighbor mean (or h0pre) ----
  gemm(xh, w1p, t, nullptr, N_NODES, F_IN, F_H);
  k_self_scale<<<grid1(NH), blk, 0, stream>>>(t, dinv, agg, NH);
  k_scatter_gcn<<<grid1(E32), blk, 0, stream>>>(row, col, dinv, t, agg, E32);
  k_bias_relu<<<grid1(NH), blk, 0, stream>>>(agg, b1, t, NH);       // h0pre -> t
  k_fill<<<grid1(NH), blk, 0, stream>>>(agg, 0.0f, NH);
  k_scatter_mean<<<grid1(E32), blk, 0, stream>>>(row, col, t, agg, E32);
  k_combine0<<<grid1(NH), blk, 0, stream>>>(t, agg, odeg, hh, jk, wsm, NH);

  // ---- layer 1 (EMA is identity in forward) ----
  gemm(hh, w2p, t, nullptr, N_NODES, F_H, F_H);
  k_self_scale<<<grid1(NH), blk, 0, stream>>>(t, dinv, agg, NH);
  k_scatter_gcn<<<grid1(E32), blk, 0, stream>>>(row, col, dinv, t, agg, E32);
  k_combine<<<grid1(NH), blk, 0, stream>>>(agg, b2, hh, jk, wsm, 1, NH);

  // ---- layer 2 ----
  gemm(hh, w3p, t, nullptr, N_NODES, F_H, F_H);
  k_self_scale<<<grid1(NH), blk, 0, stream>>>(t, dinv, agg, NH);
  k_scatter_gcn<<<grid1(E32), blk, 0, stream>>>(row, col, dinv, t, agg, E32);
  k_combine<<<grid1(NH), blk, 0, stream>>>(agg, b3, hh, jk, wsm, 2, NH);

  // ---- final: out = jk @ Wfc + bfc ----
  _Float16* jkh = (_Float16*)t;   // reuse t buffer for f16 JK features
  k_cvt_f16<<<grid1(NH), blk, 0, stream>>>(jk, jkh, NH);
  gemm(jkh, wfp, out, bfc, N_NODES, F_H, F_OUT);
}
